// CrossAttention_90615220011634
// MI455X (gfx1250) — matrix operand order repaired and checked by reading, not compiled, not run on hardware
//
#include <hip/hip_runtime.h>
#include <hip/hip_bf16.h>
#include <stdint.h>

#define NF   4
#define NS   512
#define DH   768
#define NL   103
#define NR   512
#define NACC 119
#define NTRM 11

typedef __attribute__((ext_vector_type(16))) __bf16 v16bf;
typedef __attribute__((ext_vector_type(8)))  float  v8f;

#define TM   128
#define TN   128
#define KC   64
#define APAD 8
#define ASTR (KC + APAD)        // 72 u16/row = 144B: 16B-aligned rows, conflict-free frag reads
#define TILE_U16 (TM * ASTR)    // one staging buffer
#define NSTAGE (DH / KC)        // 12 k-stages

__device__ __forceinline__ uint16_t f32_to_bf16(float f) {
    union { float f; uint32_t u; } x; x.f = f;
    uint32_t r = x.u + 0x7FFFu + ((x.u >> 16) & 1u);   // round-to-nearest-even
    return (uint16_t)(r >> 16);
}

// wave-relative LDS byte offset of a generic shared pointer (flat->LDS truncation, ISA 10.2)
__device__ __forceinline__ uint32_t lds_off_of(const void* p) {
    return (uint32_t)(uintptr_t)p;
}

// GLOBAL_LOAD_ASYNC_TO_LDS_B128, GVS mode: mem = saddr + vaddr(i32); lds = LDS_BASE + vdst
__device__ __forceinline__ void async_ld_b128(uint32_t lds_byte_off, const void* gbase,
                                              uint32_t gbyte_off) {
    asm volatile("global_load_async_to_lds_b128 %0, %1, %2"
                 :: "v"(lds_byte_off), "v"(gbyte_off), "s"(gbase)
                 : "memory");
}
template <int N>
__device__ __forceinline__ void wait_async_le() {
    asm volatile("s_wait_asynccnt %0" :: "i"(N) : "memory");
}

// stage two 128x64 bf16 tiles (A and B side) async into LDS; 8 async instrs per wave
__device__ __forceinline__ void stage_pair(uint32_t ldsA, uint32_t ldsB, const char* asrc,
                                           const char* bsrc, int kb0, int tid) {
    for (int i = tid; i < TM * 8; i += 256) {
        int r = i >> 3, c = i & 7;
        uint32_t ldsOff = (uint32_t)(r * (ASTR * 2) + c * 16);
        uint32_t gOff   = (uint32_t)(r * (DH * 2) + kb0 * 2 + c * 16);
        async_ld_b128(ldsA + ldsOff, asrc, gOff);
        async_ld_b128(ldsB + ldsOff, bsrc, gOff);
    }
}

union FragU { v16bf v; uint4 q[2]; };

// A fragment 16x32 bf16: lanes 0-15 -> M=lane, lanes 16-31 -> M=lane-16 (K offset +8)
// per-lane data = dwords [4h+kb/2 .. +3] and [8+4h+kb/2 .. +3] => two b128 LDS loads
__device__ __forceinline__ v16bf load_frag_a(const uint16_t* lds, int rowBase, int kb, int lane) {
    int m = lane & 15, h = lane >> 4;
    const uint4* p = (const uint4*)(lds + (size_t)(rowBase + m) * ASTR);
    FragU fr;
    int q0 = h + (kb >> 3);          // 16B units
    fr.q[0] = p[q0];
    fr.q[1] = p[q0 + 2];
    return fr.v;
}

// B fragment 32x16 bf16: N = lane&15, lanes 0-15 -> K 0-7,16-23; lanes 16-31 -> K 8-15,24-31
// per-lane data = dwords [4h+kb/2 .. +3] and [8+4h+kb/2 .. +3] => two b128 LDS loads
__device__ __forceinline__ v16bf load_frag_b(const uint16_t* lds, int rowBase, int kb, int lane) {
    int n = lane & 15, h = lane >> 4;
    const uint4* p = (const uint4*)(lds + (size_t)(rowBase + n) * ASTR);
    FragU fr;
    int q0 = h + (kb >> 3);          // 16B units
    fr.q[0] = p[q0];
    fr.q[1] = p[q0 + 2];
    return fr.v;
}

// ---------------------------------------------------------------- small prep
__global__ void ca_cvt_kernel(const float* __restrict__ src, uint16_t* __restrict__ dst, int n) {
    int i = blockIdx.x * blockDim.x + threadIdx.x;
    if (i < n) dst[i] = f32_to_bf16(src[i]);
}

__global__ void ca_wveff_kernel(const float* __restrict__ Wv, const float* __restrict__ w_dprob,
                                const float* __restrict__ bv, float* __restrict__ wv_eff) {
    int d = blockIdx.x * blockDim.x + threadIdx.x;
    if (d < DH) {
        float a = 0.f;
        for (int e = 0; e < DH; ++e) a += Wv[(size_t)e * DH + d] * w_dprob[e];
        wv_eff[d] = a;
    } else if (d == DH) {
        float c = 0.f;
        for (int e = 0; e < DH; ++e) c += bv[e] * w_dprob[e];
        wv_eff[DH] = c;
    }
}

// one wave32 per (l,r) row: vscore[l,r] = law_embs[l,r,:]·wv_eff + bv·w_dprob
__global__ void ca_vscore_kernel(const float* __restrict__ law_embs,
                                 const float* __restrict__ wv_eff,
                                 float* __restrict__ vscore) {
    int wid = blockIdx.x * 8 + (threadIdx.x >> 5);
    int lane = threadIdx.x & 31;
    const float* row = law_embs + (size_t)wid * DH;
    float a = 0.f;
    for (int d = lane; d < DH; d += 32) a += row[d] * wv_eff[d];
#pragma unroll
    for (int o = 16; o; o >>= 1) a += __shfl_xor(a, o, 32);
    if (lane == 0) vscore[wid] = a + wv_eff[DH];
}

// --------------------------------------------- projection GEMM (bf16 WMMA)
// Out[m,e] = bf16( sum_d A[m,d]*W[e,d] + bias[e] ), A bf16, W bf16
// double-buffered async global->LDS pipeline
#define PROJ_SMEM (4 * TILE_U16 * 2)
__global__ void __launch_bounds__(256)
ca_proj_gemm_kernel(const uint16_t* __restrict__ Ab, const uint16_t* __restrict__ Wb,
                    const float* __restrict__ bias, uint16_t* __restrict__ Out, int Mtotal) {
    extern __shared__ char psm[];
    uint16_t* lds_a = (uint16_t*)psm;             // [2][TILE_U16]
    uint16_t* lds_b = lds_a + 2 * TILE_U16;       // [2][TILE_U16]
    const int tid = threadIdx.x, lane = tid & 31, wave = tid >> 5;
    const int wm = (wave >> 2) & 1, wn = wave & 3;     // 2x4 wave grid: 64x32 per wave
    const int mTiles = Mtotal / TM;
    const int bm = (blockIdx.x % mTiles) * TM;
    const int bn = (blockIdx.x / mTiles) * TN;
    const uint32_t ldsA0 = lds_off_of(lds_a);
    const uint32_t ldsB0 = lds_off_of(lds_b);
    const char* asrc = (const char*)(Ab + (size_t)bm * DH);
    const char* bsrc = (const char*)(Wb + (size_t)bn * DH);

    v8f zf = {0.f, 0.f, 0.f, 0.f, 0.f, 0.f, 0.f, 0.f};
    v8f acc[4][2];
#pragma unroll
    for (int mi = 0; mi < 4; ++mi)
#pragma unroll
        for (int ni = 0; ni < 2; ++ni) acc[mi][ni] = zf;

    stage_pair(ldsA0, ldsB0, asrc, bsrc, 0, tid);          // prologue: stage 0 -> buf 0
    for (int s = 0; s < NSTAGE; ++s) {
        const int cur = s & 1;
        if (s + 1 < NSTAGE) {
            stage_pair(ldsA0 + (1 - cur) * (TILE_U16 * 2), ldsB0 + (1 - cur) * (TILE_U16 * 2),
                       asrc, bsrc, (s + 1) * KC, tid);
            wait_async_le<8>();                            // stage s complete, s+1 in flight
        } else {
            wait_async_le<0>();
        }
        __syncthreads();
        const uint16_t* la = lds_a + cur * TILE_U16;
        const uint16_t* lb = lds_b + cur * TILE_U16;
#pragma unroll
        for (int ks = 0; ks < KC; ks += 32) {
            v16bf afr[4], bfr[2];
#pragma unroll
            for (int mi = 0; mi < 4; ++mi) afr[mi] = load_frag_a(la, wm * 64 + mi * 16, ks, lane);
#pragma unroll
            for (int ni = 0; ni < 2; ++ni) bfr[ni] = load_frag_b(lb, wn * 32 + ni * 16, ks, lane);
#pragma unroll
            for (int mi = 0; mi < 4; ++mi)
#pragma unroll
                for (int ni = 0; ni < 2; ++ni)
                    acc[mi][ni] = __builtin_amdgcn_wmma_f32_16x16x32_bf16(
                        false, afr[mi], false, bfr[ni], (short)0, acc[mi][ni], false, false);
        }
        __syncthreads();                                   // reads done; buf may be re-staged
    }
    // epilogue: C layout -> N=lane&15, M = j + 8*(lane>>4)
    const int n0 = lane & 15, h = lane >> 4;
#pragma unroll
    for (int mi = 0; mi < 4; ++mi)
#pragma unroll
        for (int ni = 0; ni < 2; ++ni)
#pragma unroll
            for (int j = 0; j < 8; ++j) {
                int gm = bm + wm * 64 + mi * 16 + j + 8 * h;
                int gn = bn + wn * 32 + ni * 16 + n0;
                Out[(size_t)gm * DH + gn] = f32_to_bf16(acc[mi][ni][j] + bias[gn]);
            }
}

// ---------------- fused scores GEMM + masked online softmax + vscore reduce
// block = (f, l, 128-row s-tile); loops r-tiles of 128 over R=512.
#define SCSTR 129
#define SMEM_FLOATS (128 * SCSTR + 128 + 128 + 3 * 256 + 3 * 128)
#define ATTN_SMEM  (SMEM_FLOATS * 4 + 4 * TILE_U16 * 2)

__global__ void __launch_bounds__(256)
ca_attn_kernel(const uint16_t* __restrict__ Qb, const uint16_t* __restrict__ Kb,
               const float* __restrict__ vscore, const float* __restrict__ mask,
               const float* __restrict__ b_dprob, float* __restrict__ dbuf) {
    extern __shared__ char smem[];
    float* scoresLds = (float*)smem;                 // 128*129
    float* vs_tile   = scoresLds + 128 * SCSTR;      // 128
    float* mk_tile   = vs_tile + 128;                // 128
    float* pmax      = mk_tile + 128;                // 256
    float* psum      = pmax + 256;                   // 256
    float* pnum      = psum + 256;                   // 256
    float* run_m     = pnum + 256;                   // 128
    float* run_d     = run_m + 128;                  // 128
    float* run_n     = run_d + 128;                  // 128
    uint16_t* lds_a  = (uint16_t*)(run_n + 128);     // [2][TILE_U16]
    uint16_t* lds_b  = lds_a + 2 * TILE_U16;         // [2][TILE_U16]

    const int tid = threadIdx.x, lane = tid & 31, wave = tid >> 5;
    const int wm = (wave >> 2) & 1, wn = wave & 3;
    const int b = blockIdx.x;
    const int stile = b & 3, fl = b >> 2;
    const int f = fl / NL, l = fl % NL;
    const float SCALE = 0.03608439182435161f;        // 1/sqrt(768)

    const char* qsrc = (const char*)(Qb + (size_t)(f * NS + stile * 128) * DH);
    const uint16_t* Kbase = Kb + (size_t)l * NR * DH;
    const uint32_t ldsA0 = lds_off_of(lds_a);
    const uint32_t ldsB0 = lds_off_of(lds_b);

    if (tid < 128) { run_m[tid] = -3.0e38f; run_d[tid] = 0.f; run_n[tid] = 0.f; }

    v8f zf = {0.f, 0.f, 0.f, 0.f, 0.f, 0.f, 0.f, 0.f};

    for (int rt = 0; rt < 4; ++rt) {
        const int rbase = rt * 128;
        const char* ksrc = (const char*)(Kbase + (size_t)rbase * DH);
        v8f acc[4][2];
#pragma unroll
        for (int mi = 0; mi < 4; ++mi)
#pragma unroll
            for (int ni = 0; ni < 2; ++ni) acc[mi][ni] = zf;
        if (tid < 128) {
            vs_tile[tid] = vscore[(size_t)l * NR + rbase + tid];
            mk_tile[tid] = mask[(size_t)l * NR + rbase + tid];
        }
        stage_pair(ldsA0, ldsB0, qsrc, ksrc, 0, tid);          // prologue
        for (int s = 0; s < NSTAGE; ++s) {
            const int cur = s & 1;
            if (s + 1 < NSTAGE) {
                stage_pair(ldsA0 + (1 - cur) * (TILE_U16 * 2), ldsB0 + (1 - cur) * (TILE_U16 * 2),
                           qsrc, ksrc, (s + 1) * KC, tid);
                wait_async_le<8>();
            } else {
                wait_async_le<0>();
            }
            __syncthreads();
            const uint16_t* la = lds_a + cur * TILE_U16;
            const uint16_t* lb = lds_b + cur * TILE_U16;
#pragma unroll
            for (int ks = 0; ks < KC; ks += 32) {
                v16bf afr[4], bfr[2];
#pragma unroll
                for (int mi = 0; mi < 4; ++mi) afr[mi] = load_frag_a(la, wm * 64 + mi * 16, ks, lane);
#pragma unroll
                for (int ni = 0; ni < 2; ++ni) bfr[ni] = load_frag_b(lb, wn * 32 + ni * 16, ks, lane);
#pragma unroll
                for (int mi = 0; mi < 4; ++mi)
#pragma unroll
                    for (int ni = 0; ni < 2; ++ni)
                        acc[mi][ni] = __builtin_amdgcn_wmma_f32_16x16x32_bf16(
                            false, afr[mi], false, bfr[ni], (short)0, acc[mi][ni], false, false);
            }
            __syncthreads();
        }
        // dump scaled+masked scores (C layout: row = M, col = N)
        {
            const int n0 = lane & 15, h = lane >> 4;
#pragma unroll
            for (int mi = 0; mi < 4; ++mi)
#pragma unroll
                for (int ni = 0; ni < 2; ++ni)
#pragma unroll
                    for (int j = 0; j < 8; ++j) {
                        int row = wm * 64 + mi * 16 + j + 8 * h;
                        int col = wn * 32 + ni * 16 + n0;
                        scoresLds[row * SCSTR + col] = acc[mi][ni][j] * SCALE + mk_tile[col];
                    }
        }
        __syncthreads();
        // online softmax update: 2 threads per row, 64 cols each
        {
            const int row = tid >> 1, half = tid & 1;
            const float* sp = &scoresLds[row * SCSTR + half * 64];
            float lm = -3.0e38f;
#pragma unroll 8
            for (int c = 0; c < 64; ++c) lm = fmaxf(lm, sp[c]);
            pmax[tid] = lm;
            __syncthreads();
            const float tmax = fmaxf(pmax[2 * row], pmax[2 * row + 1]);
            const float nm = fmaxf(run_m[row], tmax);
            const float* vsp = vs_tile + half * 64;
            float ls = 0.f, ln = 0.f;
#pragma unroll 8
            for (int c = 0; c < 64; ++c) {
                float e = __expf(sp[c] - nm);
                ls += e; ln += e * vsp[c];
            }
            psum[tid] = ls; pnum[tid] = ln;
            __syncthreads();
            if (half == 0) {
                float so = __expf(run_m[row] - nm);
                run_d[row] = run_d[row] * so + psum[2 * row] + psum[2 * row + 1];
                run_n[row] = run_n[row] * so + pnum[2 * row] + pnum[2 * row + 1];
                run_m[row] = nm;
            }
            __syncthreads();
        }
    }
    if (tid < 128) {
        int s = stile * 128 + tid;
        dbuf[((size_t)(f * NL + l)) * NS + s] = run_n[tid] / run_d[tid] + b_dprob[0];
    }
}

// ------------------------------------------------------------- tail kernels
__global__ void ca_law_kernel(const float* __restrict__ dbuf, const float* __restrict__ w_lawprob,
                              const float* __restrict__ b_lawprob, float* __restrict__ out) {
    int wid = blockIdx.x * 4 + (threadIdx.x >> 5);
    int lane = threadIdx.x & 31;
    if (wid >= NF * NL) return;
    float a = 0.f;
    for (int s = lane; s < NS; s += 32) a += dbuf[(size_t)wid * NS + s] * w_lawprob[s];
#pragma unroll
    for (int o = 16; o; o >>= 1) a += __shfl_xor(a, o, 32);
    if (lane == 0) out[wid] = a + b_lawprob[0];
}

__global__ void ca_head_kernel(const float* __restrict__ law,
                               const float* __restrict__ W_accu, const float* __restrict__ b_accu,
                               const float* __restrict__ W_term, const float* __restrict__ b_term,
                               float* __restrict__ out) {
    int idx = blockIdx.x * blockDim.x + threadIdx.x;
    const int per = NACC + NTRM;
    if (idx >= NF * per) return;
    int f = idx / per, j = idx % per;
    const float* lrow = law + f * NL;
    if (j < NACC) {
        float a = b_accu[j];
        for (int i = 0; i < NL; ++i) a += lrow[i] * W_accu[j * NL + i];
        out[NF * NL + f * NACC + j] = a;
    } else {
        int jt = j - NACC;
        float a = b_term[jt];
        for (int i = 0; i < NL; ++i) a += lrow[i] * W_term[jt * NL + i];
        out[NF * NL + NF * NACC + f * NTRM + jt] = a;
    }
}

extern "C" void kernel_launch(void* const* d_in, const int* in_sizes, int n_in,
                              void* d_out, int out_size, void* d_ws, size_t ws_size,
                              hipStream_t stream) {
    const float* fact_rep  = (const float*)d_in[0];
    const float* law_embs  = (const float*)d_in[1];
    const float* Wq        = (const float*)d_in[2];
    const float* bq        = (const float*)d_in[3];
    const float* Wk        = (const float*)d_in[4];
    const float* bk        = (const float*)d_in[5];
    const float* Wv        = (const float*)d_in[6];
    const float* bv        = (const float*)d_in[7];
    const float* w_dprob   = (const float*)d_in[8];
    const float* b_dprob   = (const float*)d_in[9];
    const float* w_lawprob = (const float*)d_in[10];
    const float* b_lawprob = (const float*)d_in[11];
    const float* W_accu    = (const float*)d_in[12];
    const float* b_accu    = (const float*)d_in[13];
    const float* W_term    = (const float*)d_in[14];
    const float* b_term    = (const float*)d_in[15];
    const float* law_masks = (const float*)d_in[16];
    float* out = (float*)d_out;

    char* ws = (char*)d_ws;
    const size_t W_B   = (size_t)DH * DH * 2;        // bf16 weight matrix
    const size_t F_B   = (size_t)NF * NS * DH * 2;   // bf16 fact / Q
    const size_t L_B   = (size_t)NL * NR * DH * 2;   // bf16 law / K
    const size_t VS_B  = (size_t)NL * NR * 4;
    const size_t WV_B  = 772 * 4;
    uint16_t* Wqb    = (uint16_t*)(ws);
    uint16_t* Wkb    = (uint16_t*)(ws + W_B);
    uint16_t* Fb     = (uint16_t*)(ws + 2 * W_B);
    uint16_t* Lb     = (uint16_t*)(ws + 2 * W_B + F_B);
    uint16_t* Qb     = (uint16_t*)(ws + 2 * W_B + F_B + L_B);
    uint16_t* Kb     = (uint16_t*)(ws + 2 * W_B + 2 * F_B + L_B);
    float*    vscore = (float*)(ws + 2 * W_B + 2 * F_B + 2 * L_B);
    float*    wv_eff = (float*)(ws + 2 * W_B + 2 * F_B + 2 * L_B + VS_B);
    float*    dbuf   = (float*)(ws + 2 * W_B + 2 * F_B + 2 * L_B + VS_B + WV_B);

    hipFuncSetAttribute((const void*)ca_attn_kernel,
                        hipFuncAttributeMaxDynamicSharedMemorySize, ATTN_SMEM);
    hipFuncSetAttribute((const void*)ca_proj_gemm_kernel,
                        hipFuncAttributeMaxDynamicSharedMemorySize, PROJ_SMEM);

    // bf16 conversions (bandwidth-trivial prep)
    ca_cvt_kernel<<<(DH * DH) / 256, 256, 0, stream>>>(Wq, Wqb, DH * DH);
    ca_cvt_kernel<<<(DH * DH) / 256, 256, 0, stream>>>(Wk, Wkb, DH * DH);
    ca_cvt_kernel<<<(NF * NS * DH) / 256, 256, 0, stream>>>(fact_rep, Fb, NF * NS * DH);
    ca_cvt_kernel<<<(NL * NR * DH) / 256, 256, 0, stream>>>(law_embs, Lb, NL * NR * DH);
    ca_wveff_kernel<<<4, 256, 0, stream>>>(Wv, w_dprob, bv, wv_eff);
    ca_vscore_kernel<<<(NL * NR) / 8, 256, 0, stream>>>(law_embs, wv_eff, vscore);
    // Q projection: M=2048 -> 16 m-tiles x 6 n-tiles
    ca_proj_gemm_kernel<<<16 * 6, 256, PROJ_SMEM, stream>>>(Fb, Wqb, bq, Qb, NF * NS);
    // K projection: M=52736 -> 412 x 6
    ca_proj_gemm_kernel<<<412 * 6, 256, PROJ_SMEM, stream>>>(Lb, Wkb, bk, Kb, NL * NR);
    // fused attention: F*L*(S/128) = 1648 blocks
    ca_attn_kernel<<<NF * NL * 4, 256, ATTN_SMEM, stream>>>(Qb, Kb, vscore, law_masks,
                                                            b_dprob, dbuf);
    ca_law_kernel<<<NF * NL / 4, 128, 0, stream>>>(dbuf, w_lawprob, b_lawprob, out);
    ca_head_kernel<<<3, 256, 0, stream>>>(out, W_accu, b_accu, W_term, b_term, out);
    (void)in_sizes; (void)n_in; (void)out_size; (void)ws_size;
}
